// plgMultiHeadAttention_81123342287381
// MI455X (gfx1250) — compile-verified, matrix operand order checked by reading, not run on hardware
//
#include <hip/hip_runtime.h>

// ---------------------------------------------------------------------------
// Shapes (match reference): B=1, S=2048, D=2048, H=16, DEPTH=128
// ---------------------------------------------------------------------------
#define S_LEN 2048
#define D_MODEL 2048
#define N_HEADS 16
#define DEPTH 128

typedef __attribute__((ext_vector_type(16))) __bf16 v16bf;
typedef __attribute__((ext_vector_type(8)))  __bf16 v8bf;
typedef __attribute__((ext_vector_type(8)))  float  v8f;

#define CAT16(lo, hi) __builtin_shufflevector((lo), (hi), 0,1,2,3,4,5,6,7,8,9,10,11,12,13,14,15)

__device__ __forceinline__ __bf16 to_bf16(float f) {
  unsigned u = __builtin_bit_cast(unsigned, f);
  u += 0x7FFFu + ((u >> 16) & 1u);          // round-to-nearest-even
  unsigned short h = (unsigned short)(u >> 16);
  return __builtin_bit_cast(__bf16, h);
}

__device__ __forceinline__ v8f wmma_bf16(v16bf a, v16bf b, v8f c) {
  // (neg_a, A, neg_b, B, c_mod, C, reuse_a, reuse_b) -> v_wmma_f32_16x16x32_bf16
  return __builtin_amdgcn_wmma_f32_16x16x32_bf16(false, a, false, b, (short)0, c, false, false);
}

// ---------------------------------------------------------------------------
// Elementwise fp32 -> bf16
// ---------------------------------------------------------------------------
__global__ void cvt_f32_to_bf16(const float* __restrict__ in, __bf16* __restrict__ out, int n) {
  int i = blockIdx.x * blockDim.x + threadIdx.x;
  int stride = gridDim.x * blockDim.x;
  for (; i < n; i += stride) out[i] = to_bf16(in[i]);
}

// G[h][d][e] (fp32) -> Gt[h][e][d] (bf16), so qg = qh . Gt^T
__global__ void cvt_G_transpose(const float* __restrict__ G, __bf16* __restrict__ Gt) {
  int n = N_HEADS * DEPTH * DEPTH;
  int i = blockIdx.x * blockDim.x + threadIdx.x;
  int stride = gridDim.x * blockDim.x;
  for (; i < n; i += stride) {
    int h = i / (DEPTH * DEPTH);
    int r = i - h * (DEPTH * DEPTH);
    int d = r / DEPTH;
    int e = r - d * DEPTH;
    Gt[(h * DEPTH + e) * DEPTH + d] = to_bf16(G[i]);
  }
}

// ---------------------------------------------------------------------------
// Batched WMMA GEMM: C[b] = A[b] (MxK, row major) * B[b]^T (NxK, row major) + bias
// block = 256 threads = 8 waves; wave tile 64x32 (4 A-frags x 2 B-frags -> 8 WMMA
// per 32-wide k-step); waves arranged 2x4 -> block tile 128x128.
// ---------------------------------------------------------------------------
__global__ __launch_bounds__(256)
void gemm_bf16_ABt(const __bf16* __restrict__ A, long long strideA,
                   const __bf16* __restrict__ B, long long strideB,
                   float* __restrict__ C, long long strideC,
                   const float* __restrict__ bias,
                   int M, int N, int K) {
  const int wave = threadIdx.x >> 5;
  const int lane = threadIdx.x & 31;
  const int half = lane >> 4;
  const int lm   = lane & 15;

  const int m0 = blockIdx.y * 128 + (wave >> 2) * 64;   // 64 rows per wave
  const int n0 = blockIdx.x * 128 + (wave & 3) * 32;    // 32 cols per wave
  if (m0 >= M || n0 >= N) return;

  A += (long long)blockIdx.z * strideA;
  B += (long long)blockIdx.z * strideB;
  C += (long long)blockIdx.z * strideC;

  const __bf16* arow[4];
  const __bf16* brow[2];
#pragma unroll
  for (int i = 0; i < 4; ++i) arow[i] = A + (long long)(m0 + i * 16 + lm) * K;
#pragma unroll
  for (int j = 0; j < 2; ++j) brow[j] = B + (long long)(n0 + j * 16 + lm) * K;

  v8f acc[4][2];
#pragma unroll
  for (int i = 0; i < 4; ++i)
#pragma unroll
    for (int j = 0; j < 2; ++j) acc[i][j] = (v8f){};

  for (int k = 0; k < K; k += 32) {
    v16bf a[4], b[2];
#pragma unroll
    for (int i = 0; i < 4; ++i) {
      v8bf lo = *(const v8bf*)(arow[i] + k + half * 8);
      v8bf hi = *(const v8bf*)(arow[i] + k + 16 + half * 8);
      a[i] = CAT16(lo, hi);
    }
#pragma unroll
    for (int j = 0; j < 2; ++j) {
      v8bf lo = *(const v8bf*)(brow[j] + k + half * 8);
      v8bf hi = *(const v8bf*)(brow[j] + k + 16 + half * 8);
      b[j] = CAT16(lo, hi);
    }
    // speculative prefetch of the tile after next (global_prefetch_b8)
#pragma unroll
    for (int i = 0; i < 4; ++i) __builtin_prefetch((const void*)(arow[i] + k + 64), 0, 0);
#pragma unroll
    for (int j = 0; j < 2; ++j) __builtin_prefetch((const void*)(brow[j] + k + 64), 0, 0);
#pragma unroll
    for (int i = 0; i < 4; ++i)
#pragma unroll
      for (int j = 0; j < 2; ++j) acc[i][j] = wmma_bf16(a[i], b[j], acc[i][j]);
  }

#pragma unroll
  for (int j = 0; j < 2; ++j) {
    const float bv = bias ? bias[n0 + j * 16 + lm] : 0.0f;
#pragma unroll
    for (int i = 0; i < 4; ++i) {
#pragma unroll
      for (int r = 0; r < 8; ++r) {
        int m = m0 + i * 16 + r + 8 * half;
        C[(long long)m * N + n0 + j * 16 + lm] = acc[i][j][r] + bv;
      }
    }
  }
}

// ---------------------------------------------------------------------------
// RoPE + reshape: proj fp32 [S][D] -> out bf16 [H][S][DEPTH]
// ---------------------------------------------------------------------------
__global__ void rope_to_heads(const float* __restrict__ proj, __bf16* __restrict__ out) {
  const int n = S_LEN * N_HEADS * (DEPTH / 2);
  int i = blockIdx.x * blockDim.x + threadIdx.x;
  int stride = gridDim.x * blockDim.x;
  const float ln_base_scale = 9.210340371976184f / (float)DEPTH;  // ln(10000)/128
  for (; i < n; i += stride) {
    int s = i / (N_HEADS * (DEPTH / 2));
    int r = i - s * (N_HEADS * (DEPTH / 2));
    int h = r / (DEPTH / 2);
    int p = r - h * (DEPTH / 2);
    float inv_freq = __expf(-(float)(2 * p) * ln_base_scale);
    float ang = (float)s * inv_freq;
    float sn, cs;
    __sincosf(ang, &sn, &cs);
    const float* src = proj + (long long)s * D_MODEL + h * DEPTH + 2 * p;
    float xr = src[0], xi = src[1];
    __bf16* dst = out + ((long long)h * S_LEN + s) * DEPTH + 2 * p;
    dst[0] = to_bf16(xr * cs - xi * sn);
    dst[1] = to_bf16(xr * sn + xi * cs);
  }
}

// proj fp32 [S][D] -> vt bf16 [H][DEPTH][S]  (keys along the fast axis for P.V)
__global__ void v_to_heads_t(const float* __restrict__ proj, __bf16* __restrict__ vt) {
  const int n = S_LEN * D_MODEL;
  int i = blockIdx.x * blockDim.x + threadIdx.x;
  int stride = gridDim.x * blockDim.x;
  for (; i < n; i += stride) {
    int s = i / D_MODEL;
    int c = i - s * D_MODEL;
    int h = c >> 7;
    int d = c & 127;
    vt[((long long)h * DEPTH + d) * S_LEN + s] = to_bf16(proj[i]);
  }
}

// ---------------------------------------------------------------------------
// Causal flash attention. 1 wave = 16 query rows; key tiles of 32.
// qg, kh: bf16 [H][S][128]; vt: bf16 [H][128][S]; hn: bf16 [S][D]
// ---------------------------------------------------------------------------
__global__ __launch_bounds__(256)
void attn_causal(const __bf16* __restrict__ qg, const __bf16* __restrict__ kh,
                 const __bf16* __restrict__ vt, __bf16* __restrict__ hn) {
  __shared__ __bf16 lds_p[8][16 * 32];

  const int wave = threadIdx.x >> 5;
  const int lane = threadIdx.x & 31;
  const int half = lane >> 4;
  const int lm   = lane & 15;
  const int h    = blockIdx.y;
  const int qt   = blockIdx.x * 8 + wave;            // 16-row query tile

  const __bf16* Q  = qg + ((long long)h * S_LEN + qt * 16) * DEPTH;
  const __bf16* Kp = kh + (long long)h * S_LEN * DEPTH;
  const __bf16* Vt = vt + (long long)h * DEPTH * S_LEN;

  // Q fragments: 4 chunks over DEPTH=128
  v16bf qf[4];
#pragma unroll
  for (int c = 0; c < 4; ++c) {
    const __bf16* row = Q + (long long)lm * DEPTH + c * 32;
    v8bf lo = *(const v8bf*)(row + half * 8);
    v8bf hi = *(const v8bf*)(row + 16 + half * 8);
    qf[c] = CAT16(lo, hi);
  }

  float mrow[8], lrow[8];
  v8f O[8];
#pragma unroll
  for (int r = 0; r < 8; ++r) { mrow[r] = -3.0e38f; lrow[r] = 0.0f; }
#pragma unroll
  for (int c = 0; c < 8; ++c) O[c] = (v8f){};

  const float scale = 0.08838834764831845f;  // 1/sqrt(128)
  const int kt_max = (qt * 16 + 15) / 32;    // causal bound on 32-key tiles

  for (int kt = 0; kt <= kt_max; ++kt) {
    // --- scores S[16x32] = Qg . Kh^T: two 16-col halves, 4 depth chunks each
    v8f sc[2];
    sc[0] = (v8f){}; sc[1] = (v8f){};
#pragma unroll
    for (int j = 0; j < 2; ++j) {
      const __bf16* krow = Kp + (long long)(kt * 32 + j * 16 + lm) * DEPTH;
#pragma unroll
      for (int c = 0; c < 4; ++c) {
        v8bf lo = *(const v8bf*)(krow + c * 32 + half * 8);
        v8bf hi = *(const v8bf*)(krow + c * 32 + 16 + half * 8);
        v16bf b = CAT16(lo, hi);
        sc[j] = wmma_bf16(qf[c], b, sc[j]);
      }
    }

    // --- scale + causal mask (D layout: row M=r+8*half, col N=j*16+lm)
#pragma unroll
    for (int r = 0; r < 8; ++r) {
      int gq = qt * 16 + r + 8 * half;
#pragma unroll
      for (int j = 0; j < 2; ++j) {
        int gk = kt * 32 + j * 16 + lm;
        float v = sc[j][r] * scale;
        sc[j][r] = (gk > gq) ? -1.0e30f : v;
      }
    }

    // --- online softmax (row reductions across the 16 lanes of each half)
    float alpha[8];
#pragma unroll
    for (int r = 0; r < 8; ++r) {
      float rm = fmaxf(sc[0][r], sc[1][r]);
      rm = fmaxf(rm, __shfl_xor(rm, 1, 32));
      rm = fmaxf(rm, __shfl_xor(rm, 2, 32));
      rm = fmaxf(rm, __shfl_xor(rm, 4, 32));
      rm = fmaxf(rm, __shfl_xor(rm, 8, 32));
      float mn = fmaxf(mrow[r], rm);
      alpha[r] = __expf(mrow[r] - mn);
      mrow[r] = mn;
      float p0 = __expf(sc[0][r] - mn);
      float p1 = __expf(sc[1][r] - mn);
      sc[0][r] = p0; sc[1][r] = p1;
      float t = p0 + p1;
      t += __shfl_xor(t, 1, 32);
      t += __shfl_xor(t, 2, 32);
      t += __shfl_xor(t, 4, 32);
      t += __shfl_xor(t, 8, 32);
      lrow[r] = lrow[r] * alpha[r] + t;
    }

    // --- rescale O
#pragma unroll
    for (int c = 0; c < 8; ++c)
#pragma unroll
      for (int r = 0; r < 8; ++r) O[c][r] *= alpha[r];

    // --- transpose P (D layout) -> A-fragment layout through LDS
    __bf16* P = lds_p[wave];
#pragma unroll
    for (int r = 0; r < 8; ++r) {
      int row = r + 8 * half;
      P[row * 32 + lm]      = to_bf16(sc[0][r]);
      P[row * 32 + 16 + lm] = to_bf16(sc[1][r]);
    }
    asm volatile("s_wait_dscnt 0" ::: "memory");
    v8bf plo = *(const v8bf*)(P + lm * 32 + half * 8);
    v8bf phi = *(const v8bf*)(P + lm * 32 + 16 + half * 8);
    v16bf pa = CAT16(plo, phi);

    // --- O += P . V  (B columns = depth, K = keys; vt keys contiguous)
#pragma unroll
    for (int c = 0; c < 8; ++c) {
      const __bf16* vrow = Vt + (long long)(c * 16 + lm) * S_LEN + kt * 32;
      v8bf lo = *(const v8bf*)(vrow + half * 8);
      v8bf hi = *(const v8bf*)(vrow + 16 + half * 8);
      v16bf vb = CAT16(lo, hi);
      O[c] = wmma_bf16(pa, vb, O[c]);
    }
  }

  // --- normalize and store hn[S][D] as bf16 for the output projection
#pragma unroll
  for (int r = 0; r < 8; ++r) {
    float inv_l = 1.0f / lrow[r];
    long long row = (long long)(qt * 16 + r + 8 * half) * D_MODEL + h * DEPTH;
#pragma unroll
    for (int c = 0; c < 8; ++c)
      hn[row + c * 16 + lm] = to_bf16(O[c][r] * inv_l);
  }
}

// ---------------------------------------------------------------------------
// Host-side orchestration
// ---------------------------------------------------------------------------
extern "C" void kernel_launch(void* const* d_in, const int* in_sizes, int n_in,
                              void* d_out, int out_size, void* d_ws, size_t ws_size,
                              hipStream_t stream) {
  const float* q    = (const float*)d_in[0];
  const float* k    = (const float*)d_in[1];
  const float* v    = (const float*)d_in[2];
  // d_in[3] = mask (causal, applied analytically)
  const float* Wq   = (const float*)d_in[4];
  const float* bq   = (const float*)d_in[5];
  const float* Wk   = (const float*)d_in[6];
  const float* bk   = (const float*)d_in[7];
  const float* Wv   = (const float*)d_in[8];
  const float* bv   = (const float*)d_in[9];
  const float* G    = (const float*)d_in[10];
  const float* Wo   = (const float*)d_in[11];
  const float* bo   = (const float*)d_in[12];
  float* out        = (float*)d_out;

  char* ws = (char*)d_ws;
  const size_t MB = 1u << 20;
  __bf16* xbf = (__bf16*)(ws);             // 8MB  input bf16 (reused; later hn)
  __bf16* wbf = (__bf16*)(ws + 8 * MB);    // 8MB  weight bf16 (reused)
  float*  pf  = (float*) (ws + 16 * MB);   // 16MB fp32 projection (reused)
  __bf16* qh  = (__bf16*)(ws + 32 * MB);   // 8MB  [H][S][128]
  __bf16* kh  = (__bf16*)(ws + 40 * MB);   // 8MB  [H][S][128]
  __bf16* vt  = (__bf16*)(ws + 48 * MB);   // 8MB  [H][128][S]
  __bf16* qg  = (__bf16*)(ws + 56 * MB);   // 8MB  [H][S][128]
  __bf16* gt  = (__bf16*)(ws + 64 * MB);   // 0.5MB [H][128][128] (G^T)

  const int SD = S_LEN * D_MODEL;          // 4M elements
  const dim3 blk(256);
  const dim3 cgrid(2048);                  // grid-stride converters
  const dim3 gproj(D_MODEL / 128, S_LEN / 128, 1);
  const dim3 gqg(DEPTH / 128, S_LEN / 128, N_HEADS);
  const dim3 gattn(S_LEN / (16 * 8), N_HEADS);

  // ---- Q projection + RoPE
  cvt_f32_to_bf16<<<cgrid, blk, 0, stream>>>(q, xbf, SD);
  cvt_f32_to_bf16<<<cgrid, blk, 0, stream>>>(Wq, wbf, SD);
  gemm_bf16_ABt<<<gproj, blk, 0, stream>>>(xbf, 0, wbf, 0, pf, 0, bq,
                                           S_LEN, D_MODEL, D_MODEL);
  rope_to_heads<<<cgrid, blk, 0, stream>>>(pf, qh);

  // ---- K projection + RoPE
  cvt_f32_to_bf16<<<cgrid, blk, 0, stream>>>(k, xbf, SD);
  cvt_f32_to_bf16<<<cgrid, blk, 0, stream>>>(Wk, wbf, SD);
  gemm_bf16_ABt<<<gproj, blk, 0, stream>>>(xbf, 0, wbf, 0, pf, 0, bk,
                                           S_LEN, D_MODEL, D_MODEL);
  rope_to_heads<<<cgrid, blk, 0, stream>>>(pf, kh);

  // ---- V projection + transpose to [H][128][S]
  cvt_f32_to_bf16<<<cgrid, blk, 0, stream>>>(v, xbf, SD);
  cvt_f32_to_bf16<<<cgrid, blk, 0, stream>>>(Wv, wbf, SD);
  gemm_bf16_ABt<<<gproj, blk, 0, stream>>>(xbf, 0, wbf, 0, pf, 0, bv,
                                           S_LEN, D_MODEL, D_MODEL);
  v_to_heads_t<<<cgrid, blk, 0, stream>>>(pf, vt);

  // ---- qg = qh . G (per head, via Gt^T); fp32 into pf then quantize
  cvt_G_transpose<<<cgrid, blk, 0, stream>>>(G, gt);
  gemm_bf16_ABt<<<gqg, blk, 0, stream>>>(qh, (long long)S_LEN * DEPTH,
                                         gt, (long long)DEPTH * DEPTH,
                                         pf, (long long)S_LEN * DEPTH, nullptr,
                                         S_LEN, DEPTH, DEPTH);
  cvt_f32_to_bf16<<<cgrid, blk, 0, stream>>>(pf, qg, SD);

  // ---- causal flash attention -> hn bf16 [S][D] (reuses xbf)
  attn_causal<<<gattn, blk, 0, stream>>>(qg, kh, vt, xbf);

  // ---- output projection: out = hn . Wo^T + bo (fp32)
  cvt_f32_to_bf16<<<cgrid, blk, 0, stream>>>(Wo, wbf, SD);
  gemm_bf16_ABt<<<gproj, blk, 0, stream>>>(xbf, 0, wbf, 0, out, 0, bo,
                                           S_LEN, D_MODEL, D_MODEL);
}
